// RegionalMultiHeadAttention_20529943675115
// MI455X (gfx1250) — compile-verified, matrix operand order checked by reading, not run on hardware
//
#include <hip/hip_runtime.h>
#include <hip/hip_bf16.h>

// Problem constants (match reference)
constexpr int Bc = 2, Sc = 32, Cc = 256, Lc = 640, Hc = 5, Dc = 128;
constexpr int Nrows = Bc * Sc * Cc;   // 16384 total (b,s,c) rows
constexpr int QKVN  = 3 * Hc * Dc;    // 1920 fused q|k|v output features

typedef __attribute__((ext_vector_type(16))) __bf16 v16bf;
typedef __attribute__((ext_vector_type(8)))  __bf16 v8bf;
typedef __attribute__((ext_vector_type(8)))  float  v8f;
typedef __attribute__((ext_vector_type(4)))  unsigned int u32x4;
typedef __attribute__((ext_vector_type(8)))  int i32x8;
typedef __attribute__((ext_vector_type(4)))  int i32x4;

// ---------------------------------------------------------------------------
// Load a 16x32 bf16 WMMA fragment (A or B^T form) from K-contiguous memory.
// ISA 16-bit A layout: lanes 0-15 hold row=lane, K = {kh*8..+7, 16+kh*8..+7},
// kh = lane>>4. Both halves are 8 contiguous bf16 -> two b128 loads.
// Works for global or LDS pointers (generic addressing).
// ---------------------------------------------------------------------------
__device__ inline v16bf load_frag(const __bf16* p0, int ld) {
  const int lane = threadIdx.x & 31;
  const int r    = lane & 15;
  const int kh   = lane >> 4;
  const __bf16* p = p0 + (size_t)r * ld + kh * 8;
  v8bf lo = *(const v8bf*)(p);
  v8bf hi = *(const v8bf*)(p + 16);
  v16bf out;
#pragma unroll
  for (int i = 0; i < 8; ++i) { out[i] = lo[i]; out[i + 8] = hi[i]; }
  return out;
}

__device__ inline v8f wmma_bf16(v16bf a, v16bf b, v8f c) {
  // D = A(16x32) * B(32x16) + C, fp32 accumulate
  return __builtin_amdgcn_wmma_f32_16x16x32_bf16(
      /*neg_a=*/false, a, /*neg_b=*/false, b,
      /*c_mod=*/(short)0, c, /*reuse_a=*/false, /*reuse_b=*/false);
}

// ---------------------------------------------------------------------------
// TDM: async 2-D bf16 tile load (tile_rows x tile_cols) from a row-major
// [rows_total x K] matrix into LDS at byte offset lds_off, with hardware LDS
// padding: 16 B inserted after every 128 B row (pad_interval=4 -> 32 DWORDs,
// pad_amount=3 -> 4 DWORDs), i.e. padded LDS pitch = 144 B (72 elements).
// D# layout per CDNA5 ISA 8.3/8.4. Tracked by TENSORcnt.
// ---------------------------------------------------------------------------
__device__ inline void tdm_load_2d(unsigned lds_off, const __bf16* gptr,
                                   int rows_total, int K,
                                   int tile_rows, int tile_cols) {
  const unsigned long long ga = (unsigned long long)(size_t)gptr;
  u32x4 g0;
  g0[0] = 1u;                                            // count=1, user mode
  g0[1] = lds_off;                                       // lds_addr (bytes)
  g0[2] = (unsigned)(ga & 0xffffffffu);                  // global_addr[31:0]
  g0[3] = (unsigned)((ga >> 32) & 0x01ffffffu)           // global_addr[56:32]
          | (2u << 30);                                  // type=2 ("image")
  i32x8 g1;
  g1[0] = (int)((1u << 16)        // data_size=1 -> 2 bytes
                | (1u << 20)      // pad_enable
                | (4u << 22)      // pad_interval: 32 DWORDs (one 128B row)
                | (3u << 25));    // pad_amount: 4 DWORDs (16 B)
  g1[1] = (int)(((unsigned)K & 0xffffu) << 16);                         // tensor_dim0[15:0]
  g1[2] = (int)(((unsigned)K >> 16) |
                (((unsigned)rows_total & 0xffffu) << 16));              // dim0 hi | dim1 lo
  g1[3] = (int)(((unsigned)rows_total >> 16) |
                ((unsigned)tile_cols << 16));                           // dim1 hi | tile_dim0
  g1[4] = (int)((unsigned)tile_rows);                                   // tile_dim1 | tile_dim2=0
  g1[5] = (int)((unsigned)K);                                           // dim0_stride[31:0]
  g1[6] = 0;                                                            // stride hi | dim1_stride lo
  g1[7] = 0;
  const i32x4 z4 = {0, 0, 0, 0};
#if __has_include(<hip/amd_detail/amd_gfx1250_TDM.h>)
  const i32x8 z8 = {0, 0, 0, 0, 0, 0, 0, 0};
  __builtin_amdgcn_tensor_load_to_lds(g0, g1, z4, z4, z8, 0);   // clang-23 form
#else
  __builtin_amdgcn_tensor_load_to_lds(g0, g1, z4, z4, 0);       // ROCm 7.2 form
#endif
}

// ---------------------------------------------------------------------------
// fp32 -> bf16 conversion
// ---------------------------------------------------------------------------
__global__ __launch_bounds__(256) void cvt_f32_bf16(const float* __restrict__ s,
                                                    __bf16* __restrict__ d, int n) {
  int i = blockIdx.x * 256 + threadIdx.x;
  if (i < n) d[i] = (__bf16)s[i];
}

// ---------------------------------------------------------------------------
// GEMM: out[m][n] = sum_k A[m][k] * W[n][k]  (+ bias), A,W bf16, K-contiguous.
// Block = 128x128 tile, 256 threads = 8 wave32 (2x4); wave tile = 64x32.
// K is consumed in 64-wide slabs, TDM-staged into double-buffered LDS
// (padded pitch 72 elems), prefetching slab kt+1 while computing slab kt.
// ---------------------------------------------------------------------------
constexpr int BK     = 64;         // K-slab
constexpr int LDP    = 72;         // padded LDS row pitch (elements)
constexpr int TILE_E = 128 * LDP;  // 9216 elements per staged tile
constexpr int GEMM_SMEM_BYTES = 4 * TILE_E * (int)sizeof(__bf16);  // 73728

__global__ __launch_bounds__(256)
void gemm_bf16_kernel(const __bf16* __restrict__ A, const __bf16* __restrict__ W,
                      int M, int K, int Ntot,
                      __bf16* __restrict__ obf, float* __restrict__ of32,
                      const float* __restrict__ bias) {
  extern __shared__ __bf16 gsmem[];   // [buf0:A,B][buf1:A,B], 72-elem pitch
  const int tid  = threadIdx.x;
  const int wave = tid >> 5;
  const int lane = tid & 31;
  const int wm   = wave >> 2;                 // 0..1
  const int wn   = wave & 3;                  // 0..3
  const int m0   = blockIdx.x * 128;
  const int n0   = blockIdx.y * 128;

  auto issue = [&](int buf, int k0) {
    if (wave == 0) {
      const unsigned aoff = (unsigned)(buf * 2 * TILE_E * sizeof(__bf16));
      const unsigned boff = aoff + (unsigned)(TILE_E * sizeof(__bf16));
      tdm_load_2d(aoff, A + (size_t)m0 * K + k0, M,    K, 128, BK);
      tdm_load_2d(boff, W + (size_t)n0 * K + k0, Ntot, K, 128, BK);
    }
  };

  const int nk = K / BK;
  v8f acc[4][2] = {};
  issue(0, 0);
  for (int kt = 0; kt < nk; ++kt) {
    const int buf = kt & 1;
    if (kt + 1 < nk) issue(buf ^ 1, (kt + 1) * BK);
    if (wave == 0) {
      if (kt + 1 < nk) __builtin_amdgcn_s_wait_tensorcnt(2);  // slab kt landed
      else             __builtin_amdgcn_s_wait_tensorcnt(0);
    }
    __syncthreads();

    const __bf16* tA = gsmem + buf * 2 * TILE_E;
    const __bf16* tB = tA + TILE_E;
#pragma unroll
    for (int ks = 0; ks < 2; ++ks) {
      v16bf a[4], b[2];
#pragma unroll
      for (int i = 0; i < 4; ++i)
        a[i] = load_frag(tA + (wm * 64 + i * 16) * LDP + ks * 32, LDP);
#pragma unroll
      for (int j = 0; j < 2; ++j)
        b[j] = load_frag(tB + (wn * 32 + j * 16) * LDP + ks * 32, LDP);
#pragma unroll
      for (int i = 0; i < 4; ++i)
#pragma unroll
        for (int j = 0; j < 2; ++j)
          acc[i][j] = wmma_bf16(a[i], b[j], acc[i][j]);
    }
    __syncthreads();   // all waves done reading buf before its next TDM issue
  }

  // C/D layout: VGPR r -> M = r (lanes 0-15) or r+8 (lanes 16-31); N = lane&15
  const int r16 = lane & 15;
  const int hi8 = (lane >> 4) * 8;
#pragma unroll
  for (int i = 0; i < 4; ++i)
#pragma unroll
    for (int j = 0; j < 2; ++j)
#pragma unroll
      for (int r = 0; r < 8; ++r) {
        const int m = m0 + wm * 64 + i * 16 + r + hi8;
        const int n = n0 + wn * 32 + j * 16 + r16;
        const float v = acc[i][j][r];
        if (of32) of32[(size_t)m * Ntot + n] = v + (bias ? bias[n] : 0.0f);
        else      obf[(size_t)m * Ntot + n]  = (__bf16)v;
      }
}

// ---------------------------------------------------------------------------
// Attention over one (b*s, h): Q,K,V are 256x128 slices of the fused Y buffer
// (row stride 1920). Scores (256x256, bf16) and V^T (128x256, bf16) live in
// 192 KB of LDS (CDNA5: up to 320 KB per workgroup). 8 waves, each owning 32
// query rows: 128 WMMAs for QK^T + 128 WMMAs for PV per wave.
// ---------------------------------------------------------------------------
__global__ __launch_bounds__(256)
void attn_kernel(const __bf16* __restrict__ Y, __bf16* __restrict__ Oc) {
  extern __shared__ __bf16 smem[];
  __bf16* sS  = smem;              // [256][256] scores -> probabilities
  __bf16* sVt = smem + Cc * Cc;    // [128][256] V^T

  const int bs = blockIdx.x / Hc;
  const int h  = blockIdx.x % Hc;
  const __bf16* Qp = Y + (size_t)bs * Cc * QKVN + h * Dc;
  const __bf16* Kp = Qp + Hc * Dc;       // +640
  const __bf16* Vp = Qp + 2 * Hc * Dc;   // +1280

  const int tid = threadIdx.x, wave = tid >> 5, lane = tid & 31;

  // Stage V^T into LDS: sVt[d][c] = V[c][d]
  for (int idx = tid; idx < Dc * Cc; idx += 256) {
    const int d = idx >> 8;        // / 256
    const int c = idx & 255;
    sVt[d * Cc + c] = Vp[(size_t)c * QKVN + d];
  }

  // Preload this wave's Q fragments: rows [wave*32, wave*32+32), K = 128
  v16bf qf[2][4];
#pragma unroll
  for (int mt = 0; mt < 2; ++mt)
#pragma unroll
    for (int ks = 0; ks < 4; ++ks)
      qf[mt][ks] = load_frag(Qp + (size_t)(wave * 32 + mt * 16) * QKVN + ks * 32, QKVN);

  const float scale = 0.08838834764831845f;  // 128^-0.5
  const int r16 = lane & 15;
  const int hi8 = (lane >> 4) * 8;

  // S = scale * Q K^T, streamed over 16 key tiles, stored bf16 into LDS
  for (int nb = 0; nb < 16; ++nb) {
    v8f acc[2] = {};
#pragma unroll
    for (int ks = 0; ks < 4; ++ks) {
      v16bf kf = load_frag(Kp + (size_t)(nb * 16) * QKVN + ks * 32, QKVN);
      acc[0] = wmma_bf16(qf[0][ks], kf, acc[0]);
      acc[1] = wmma_bf16(qf[1][ks], kf, acc[1]);
    }
#pragma unroll
    for (int mt = 0; mt < 2; ++mt)
#pragma unroll
      for (int r = 0; r < 8; ++r) {
        const int m = wave * 32 + mt * 16 + r + hi8;
        const int n = nb * 16 + r16;
        sS[m * Cc + n] = (__bf16)(acc[mt][r] * scale);
      }
  }
  __syncthreads();

  // Row softmax: one lane per row (wave owns rows wave*32 .. +31)
  {
    __bf16* sr = sS + (wave * 32 + lane) * Cc;
    float mx = -3.0e38f;
    for (int c = 0; c < Cc; ++c) mx = fmaxf(mx, (float)sr[c]);
    float sum = 0.0f;
    for (int c = 0; c < Cc; ++c) {
      const float e = __expf((float)sr[c] - mx);
      sum += e;
      sr[c] = (__bf16)e;
    }
    const float inv = 1.0f / sum;
    for (int c = 0; c < Cc; ++c) sr[c] = (__bf16)((float)sr[c] * inv);
  }
  __syncthreads();

  // O = P V  (K-dim = 256 keys, 8 k-steps); P and V^T fragments from LDS
  v8f o[2][8] = {};
  for (int ks = 0; ks < 8; ++ks) {
    v16bf pf[2];
    pf[0] = load_frag(sS + (size_t)(wave * 32) * Cc + ks * 32, Cc);
    pf[1] = load_frag(sS + (size_t)(wave * 32 + 16) * Cc + ks * 32, Cc);
#pragma unroll
    for (int dt = 0; dt < 8; ++dt) {
      v16bf vf = load_frag(sVt + (size_t)(dt * 16) * Cc + ks * 32, Cc);
      o[0][dt] = wmma_bf16(pf[0], vf, o[0][dt]);
      o[1][dt] = wmma_bf16(pf[1], vf, o[1][dt]);
    }
  }

  // Write head output into concat buffer [16384][640] at column h*128
#pragma unroll
  for (int mt = 0; mt < 2; ++mt)
#pragma unroll
    for (int dt = 0; dt < 8; ++dt)
#pragma unroll
      for (int r = 0; r < 8; ++r) {
        const int c = wave * 32 + mt * 16 + r + hi8;
        const int d = dt * 16 + r16;
        Oc[((size_t)bs * Cc + c) * Lc + h * Dc + d] = (__bf16)o[mt][dt][r];
      }
}

// ---------------------------------------------------------------------------
// Host-side orchestration
// ---------------------------------------------------------------------------
extern "C" void kernel_launch(void* const* d_in, const int* in_sizes, int n_in,
                              void* d_out, int out_size, void* d_ws, size_t ws_size,
                              hipStream_t stream) {
  const float* x  = (const float*)d_in[0];
  const float* Wq = (const float*)d_in[1];
  const float* Wk = (const float*)d_in[2];
  const float* Wv = (const float*)d_in[3];
  const float* Wp = (const float*)d_in[4];
  const float* bp = (const float*)d_in[5];
  float* out = (float*)d_out;

  char* ws = (char*)d_ws;
  size_t off = 0;
  auto carve = [&](size_t bytes) -> void* {
    void* p = ws + off;
    off += (bytes + 255) & ~(size_t)255;
    return p;
  };
  __bf16* xb   = (__bf16*)carve((size_t)Nrows * Lc * 2);   // x in bf16
  __bf16* wcat = (__bf16*)carve((size_t)QKVN * Lc * 2);    // [Wq;Wk;Wv] bf16
  __bf16* wpb  = (__bf16*)carve((size_t)Lc * Lc * 2);      // Wp bf16
  __bf16* Y    = (__bf16*)carve((size_t)Nrows * QKVN * 2); // fused q|k|v
  __bf16* Oc   = (__bf16*)carve((size_t)Nrows * Lc * 2);   // concat heads

  auto cvt = [&](const float* s, __bf16* d, int n) {
    cvt_f32_bf16<<<(n + 255) / 256, 256, 0, stream>>>(s, d, n);
  };
  cvt(x,  xb,   Nrows * Lc);
  cvt(Wq, wcat,                 Hc * Dc * Lc);
  cvt(Wk, wcat +     Hc * Dc * Lc, Hc * Dc * Lc);
  cvt(Wv, wcat + 2 * Hc * Dc * Lc, Hc * Dc * Lc);
  cvt(Wp, wpb, Lc * Lc);

  (void)hipFuncSetAttribute((const void*)gemm_bf16_kernel,
                            hipFuncAttributeMaxDynamicSharedMemorySize,
                            GEMM_SMEM_BYTES);

  // Fused QKV projection: [16384,640] x [1920,640]^T -> [16384,1920] bf16
  dim3 g1(Nrows / 128, QKVN / 128);
  gemm_bf16_kernel<<<g1, 256, GEMM_SMEM_BYTES, stream>>>(
      xb, wcat, Nrows, Lc, QKVN, Y, nullptr, nullptr);

  // Attention per (b*s, h): 320 workgroups, 192 KB dynamic LDS each
  const size_t smem = (size_t)(Cc * Cc + Dc * Cc) * sizeof(__bf16); // 196608 B
  (void)hipFuncSetAttribute((const void*)attn_kernel,
                            hipFuncAttributeMaxDynamicSharedMemorySize, (int)smem);
  attn_kernel<<<Bc * Sc * Hc, 256, smem, stream>>>(Y, Oc);

  // Output projection: [16384,640] x [640,640]^T + bias -> fp32 d_out
  dim3 g2(Nrows / 128, Lc / 128);
  gemm_bf16_kernel<<<g2, 256, GEMM_SMEM_BYTES, stream>>>(
      Oc, wpb, Nrows, Lc, Lc, nullptr, out, bp);
}
